// NoisyTopKGating_50740743635375
// MI455X (gfx1250) — matrix operand order repaired and checked by reading, not compiled, not run on hardware
//
#include <hip/hip_runtime.h>

// CDNA5 / gfx1250 WMMA types (wave32)
typedef __attribute__((ext_vector_type(16))) __bf16 v16bf;
typedef __attribute__((ext_vector_type(8)))  float  v8f;

// Problem constants (from reference): B=4, S=4096, D=2048, E=64, K=2
constexpr int Bs = 4, Ss = 4096, Ds = 2048, Es = 64;
constexpr int T  = Bs * Ss;     // 16384 tokens
constexpr int ROWS    = 64;     // tokens per workgroup (4 waves x 16 rows)
constexpr int THREADS = 128;    // 4 wave32 waves
constexpr int KC      = 32;     // K per v_wmma_f32_16x16x32_bf16
constexpr int NITER   = Ds / KC;

__global__ __launch_bounds__(THREADS)
void noisy_topk_gating(const float* __restrict__ x,
                       const float* __restrict__ Wg,
                       const float* __restrict__ bg,
                       const float* __restrict__ Wn,
                       const float* __restrict__ bn,
                       const float* __restrict__ noise,
                       float* __restrict__ probs,   // [T][E]
                       float* __restrict__ topk)    // [T][2] (indices, numeric)
{
    // double-buffered bf16 weight stage: [buf][mat][expert][k]  (2 x 8 KB)
    __shared__ __align__(32) __bf16 wlds[2][2][Es][KC];
    // epilogue transpose buffers (pad +1 float -> no 32-way bank conflicts)
    __shared__ float eg[ROWS][Es + 1];
    __shared__ float en[ROWS][Es + 1];

    const int tid  = threadIdx.x;
    const int wave = tid >> 5;
    const int lane = tid & 31;
    const int h    = lane >> 4;   // lane half (A/B: K-group, C: M-half)
    const int n16  = lane & 15;

    const int rowbase = blockIdx.x * ROWS + wave * 16;

    v8f zero;
    #pragma unroll
    for (int i = 0; i < 8; ++i) zero[i] = 0.0f;
    v8f accg[4] = {zero, zero, zero, zero};   // gate logits, 16x64 per wave
    v8f accn[4] = {zero, zero, zero, zero};   // noise logits, 16x64 per wave

    // cooperative weight staging: thread -> (matrix, expert row), 32 k each
    const int wmat = tid >> 6;          // 0 -> Wg, 1 -> Wn
    const int wn   = tid & 63;
    const float* wsrc = (wmat ? Wn : Wg) + (size_t)wn * Ds;

    // A-fragment source row for this lane (contiguous K order matching B:
    // lanes 0-15 hold K=0..15, lanes 16-31 hold K=16..31)
    const float* xrow = x + (size_t)(rowbase + n16) * Ds;

    // ---- prologue: stage chunk 0 into buffer 0 --------------------------
    {
        const float4* s = (const float4*)wsrc;
        float4 w[8];
        #pragma unroll
        for (int i = 0; i < 8; ++i) w[i] = s[i];
        __bf16* d = &wlds[0][wmat][wn][0];
        #pragma unroll
        for (int i = 0; i < 8; ++i) {
            d[4*i+0] = (__bf16)w[i].x;  d[4*i+1] = (__bf16)w[i].y;
            d[4*i+2] = (__bf16)w[i].z;  d[4*i+3] = (__bf16)w[i].w;
        }
    }
    __syncthreads();

    // ---- software-pipelined main loop: one barrier per K-chunk ----------
    int cur = 0;
    for (int it = 0; it < NITER; ++it) {
        const int kc = it * KC;
        const bool has_next = (it + 1 < NITER);

        // issue next chunk's weight loads now; latency hides under the WMMAs
        float4 wnext[8];
        if (has_next) {
            const float4* s = (const float4*)(wsrc + kc + KC);
            #pragma unroll
            for (int i = 0; i < 8; ++i) wnext[i] = s[i];
        }

        // A fragment: 16 fp32 of this lane's token row -> bf16
        v16bf a;
        {
            const float4 a0 = *(const float4*)(xrow + kc + h * 16);
            const float4 a1 = *(const float4*)(xrow + kc + h * 16 + 4);
            const float4 a2 = *(const float4*)(xrow + kc + h * 16 + 8);
            const float4 a3 = *(const float4*)(xrow + kc + h * 16 + 12);
            a[0]  = (__bf16)a0.x; a[1]  = (__bf16)a0.y;
            a[2]  = (__bf16)a0.z; a[3]  = (__bf16)a0.w;
            a[4]  = (__bf16)a1.x; a[5]  = (__bf16)a1.y;
            a[6]  = (__bf16)a1.z; a[7]  = (__bf16)a1.w;
            a[8]  = (__bf16)a2.x; a[9]  = (__bf16)a2.y;
            a[10] = (__bf16)a2.z; a[11] = (__bf16)a2.w;
            a[12] = (__bf16)a3.x; a[13] = (__bf16)a3.y;
            a[14] = (__bf16)a3.z; a[15] = (__bf16)a3.w;
        }
        if (kc + 8 * KC < Ds)
            __builtin_prefetch(xrow + kc + 8 * KC, 0, 1);  // global_prefetch_b8

        // 4 N-tiles x {gate, noise}; B fragment = 16 consecutive k of one
        // expert row (contiguous k order, same as A: lanes 0-15 K=0..15, lanes 16-31 K=16..31)
        #pragma unroll
        for (int nt = 0; nt < 4; ++nt) {
            const v16bf bgf = *(const v16bf*)&wlds[cur][0][nt * 16 + n16][h * 16];
            const v16bf bnf = *(const v16bf*)&wlds[cur][1][nt * 16 + n16][h * 16];
            accg[nt] = __builtin_amdgcn_wmma_f32_16x16x32_bf16(
                false, a, false, bgf, (short)0, accg[nt], false, false);
            accn[nt] = __builtin_amdgcn_wmma_f32_16x16x32_bf16(
                false, a, false, bnf, (short)0, accn[nt], false, false);
        }

        // convert + store the prefetched chunk into the other buffer
        if (has_next) {
            __bf16* d = &wlds[cur ^ 1][wmat][wn][0];
            #pragma unroll
            for (int i = 0; i < 8; ++i) {
                d[4*i+0] = (__bf16)wnext[i].x;  d[4*i+1] = (__bf16)wnext[i].y;
                d[4*i+2] = (__bf16)wnext[i].z;  d[4*i+3] = (__bf16)wnext[i].w;
            }
        }

        // single barrier: orders this iter's buf[cur] reads / buf[nxt] writes
        // against the next iter's buf[nxt] reads / buf[cur] writes
        __syncthreads();
        cur ^= 1;
    }

    // C/D layout: VGPR r, lane half h -> M = h*8 + r, N = n16 (per 16x16 tile)
    #pragma unroll
    for (int nt = 0; nt < 4; ++nt) {
        #pragma unroll
        for (int r = 0; r < 8; ++r) {
            const int rl = wave * 16 + h * 8 + r;
            const int c  = nt * 16 + n16;
            eg[rl][c] = accg[nt][r];
            en[rl][c] = accn[nt][r];
        }
    }
    __syncthreads();

    // one thread per token: bias + softplus noise + top-2 + scattered softmax
    if (tid < ROWS) {
        const int row   = blockIdx.x * ROWS + tid;
        const float* nz = noise + (size_t)row * Es;
        float v0 = -3.402823466e38f, v1 = -3.402823466e38f;
        int   i0 = 0, i1 = 0;
        for (int e = 0; e < Es; ++e) {
            const float g    = eg[tid][e] + bg[e];
            const float sarg = en[tid][e] + bn[e];
            const float sp   = (sarg > 20.0f) ? sarg : log1pf(__expf(sarg));
            const float v    = fmaf(nz[e], sp, g);
            if (v > v0)      { v1 = v0; i1 = i0; v0 = v; i0 = e; }
            else if (v > v1) { v1 = v;  i1 = e; }
        }
        // softmax over {-inf ... v0 ... v1 ...} == 2-way softmax at i0/i1
        const float e1 = __expf(v1 - v0);
        const float p0 = 1.0f / (1.0f + e1);
        const float p1 = e1 * p0;
        float* orow = probs + (size_t)row * Es;
        #pragma unroll 8
        for (int e = 0; e < Es; ++e) orow[e] = 0.0f;
        orow[i0] = p0;
        orow[i1] = p1;
        topk[row * 2 + 0] = (float)i0;   // int32 indices emitted numerically
        topk[row * 2 + 1] = (float)i1;
    }
}

extern "C" void kernel_launch(void* const* d_in, const int* /*in_sizes*/, int /*n_in*/,
                              void* d_out, int /*out_size*/, void* /*d_ws*/, size_t /*ws_size*/,
                              hipStream_t stream)
{
    const float* x   = (const float*)d_in[0];
    const float* Wg  = (const float*)d_in[1];
    const float* bg  = (const float*)d_in[2];
    const float* Wn  = (const float*)d_in[3];
    const float* bn  = (const float*)d_in[4];
    const float* nz  = (const float*)d_in[5];

    float* probs = (float*)d_out;                       // [T][E]
    float* topk  = (float*)d_out + (size_t)T * Es;      // [T][2]

    noisy_topk_gating<<<dim3(T / ROWS), dim3(THREADS), 0, stream>>>(
        x, Wg, bg, Wn, bn, nz, probs, topk);
}
